// FastAttention_74552042324473
// MI455X (gfx1250) — matrix operand order repaired and checked by reading, not compiled, not run on hardware
//
#include <hip/hip_runtime.h>
#include <math.h>

// ---- problem constants (match reference) ----
#define BB    2
#define SS    2048
#define DD    256
#define DQK   64
#define HH    8
#define RK    16
#define NRFF  128
#define KSEL  64
#define NBUCK 64

typedef __attribute__((ext_vector_type(16))) __bf16 v16bf;
typedef __attribute__((ext_vector_type(8)))  float  v8f;
typedef __attribute__((ext_vector_type(4)))  unsigned int u32x4;
typedef __attribute__((ext_vector_type(8)))  int i32x8;
typedef __attribute__((ext_vector_type(4)))  int i32x4;

// ============================================================
// WMMA fragment loaders (CDNA5 16-bit layouts, wave32)
// ============================================================
// A (16x32, row-major src [M,K]): lanes 0-15 -> M=lane, K kb..kb+15 with kb=0
//                                 lanes 16-31 -> kb=16
__device__ __forceinline__ v16bf frag_a(const __bf16* __restrict__ A, int lda,
                                        int m0, int k0, int lane) {
  int m  = m0 + (lane & 15);
  int kb = (lane < 16) ? 0 : 16;
  const __bf16* p = A + (size_t)m * lda + k0 + kb;
  v16bf v;
#pragma unroll
  for (int i = 0; i < 8; ++i) v[i] = p[i];
#pragma unroll
  for (int i = 0; i < 8; ++i) v[8 + i] = p[8 + i];
  return v;
}

// B (32x16) from row-major weights [K,N]: lanes 0-15 hold K=k0..k0+15, lanes 16-31 K=k0+16..+31
__device__ __forceinline__ v16bf frag_b_rm(const __bf16* __restrict__ Bm, int ldb,
                                           int n0, int k0, int lane) {
  int n  = n0 + (lane & 15);
  int kb = (lane < 16) ? 0 : 16;
  const __bf16* p = Bm + (size_t)(k0 + kb) * ldb + n;
  v16bf v;
#pragma unroll
  for (int i = 0; i < 16; ++i) v[i] = p[(size_t)i * ldb];
  return v;
}

// B fragment from an LDS tile holding X row-major [16 keys][ldb] (logical B = Xᵀ): contiguous
__device__ __forceinline__ v16bf frag_b_lds_tr(const __bf16* tile, int ldb,
                                               int k0, int lane) {
  int n  = lane & 15;
  int kb = (lane < 16) ? 0 : 16;
  const __bf16* p = tile + (size_t)n * ldb + k0 + kb;
  v16bf v;
#pragma unroll
  for (int i = 0; i < 16; ++i) v[i] = p[i];
  return v;
}

// B fragment from an LDS panel holding B row-major [K][16]
__device__ __forceinline__ v16bf frag_b_lds_rm(const __bf16* panel, int k0, int lane) {
  int n  = lane & 15;
  int kb = (lane < 16) ? 0 : 16;
  const __bf16* p = panel + (size_t)(k0 + kb) * 16 + n;
  v16bf v;
#pragma unroll
  for (int i = 0; i < 16; ++i) v[i] = p[(size_t)i * 16];
  return v;
}

__device__ __forceinline__ v8f wmma_bf16(v16bf a, v16bf b, v8f c) {
  return __builtin_amdgcn_wmma_f32_16x16x32_bf16(false, a, false, b, (short)0, c,
                                                 false, false);
}

// ============================================================
// Tensor Data Mover: DMA a 2D tile (bf16 elements) from global into LDS.
// D# per CDNA5 ISA. Groups 2/3 zero (<=2D tensor).
// clang-23 toolchain: 6-arg builtin (g0 u32x4, g1 i32x8, g2 i32x4, g3 i32x4,
// extra i32x8, cpol i32).
//   tensor_d0 = row length of tensor (elements), tensor_d1 = #rows
//   tile_d0   = tile row length (elements, *2B must be mult of 4B)
//   tile_d1   = tile #rows (0 => 1D tile)
//   stride_d0 = tensor dim0 stride (elements)
// ============================================================
__device__ __forceinline__ void tdm_load_to_lds(unsigned lds_off, const void* gptr,
                                                unsigned tensor_d0, unsigned tensor_d1,
                                                unsigned tile_d0, unsigned tile_d1,
                                                unsigned stride_d0) {
  unsigned long long ga = (unsigned long long)gptr;
  u32x4 g0;
  g0[0] = 1u;                                            // count=1 (valid descriptor)
  g0[1] = lds_off;                                       // lds_addr (bytes)
  g0[2] = (unsigned)(ga & 0xffffffffull);                // global_addr[31:0]
  g0[3] = (unsigned)((ga >> 32) & 0x1ffffffull)          // global_addr[56:32]
          | (2u << 30);                                  // type=2 ("image")
  i32x8 g1;
  g1[0] = (int)(1u << 16);                               // data_size=1 -> 2 bytes
  g1[1] = (int)((tensor_d0 & 0xffffu) << 16);            // tensor_dim0[15:0]
  g1[2] = (int)((tensor_d0 >> 16) |                      // tensor_dim0[31:16]
                ((tensor_d1 & 0xffffu) << 16));          // tensor_dim1[15:0]
  g1[3] = (int)((tensor_d1 >> 16) |                      // tensor_dim1[31:16]
                ((tile_d0 & 0xffffu) << 16));            // tile_dim0
  g1[4] = (int)(tile_d1 & 0xffffu);                      // tile_dim1 (tile_dim2=0)
  g1[5] = (int)stride_d0;                                // tensor_dim0_stride[31:0]
  g1[6] = 0;                                             // stride hi / dim1_stride lo
  g1[7] = 0;
  i32x4 gz4 = {0, 0, 0, 0};
  i32x8 gz8 = {0, 0, 0, 0, 0, 0, 0, 0};
  __builtin_amdgcn_tensor_load_to_lds(g0, g1, gz4, gz4, gz8, 0);
}

// ============================================================
// small helper kernels
// ============================================================
__global__ void k_cvt(const float* __restrict__ src, __bf16* __restrict__ dst, int n) {
  int i = blockIdx.x * blockDim.x + threadIdx.x;
  if (i < n) dst[i] = (__bf16)src[i];
}

// Wc[h][m][n] = sum_r U[h][m][r] * V[h][r][n]   (rank-16 combine), m<64
__global__ void k_combo(const float* __restrict__ U, const float* __restrict__ V,
                        __bf16* __restrict__ Wc, int N) {
  int i = blockIdx.x * blockDim.x + threadIdx.x;
  int total = HH * 64 * N;
  if (i >= total) return;
  int n = i % N;
  int rest = i / N;
  int m = rest % 64;
  int h = rest / 64;
  const float* u = U + ((size_t)h * 64 + m) * RK;
  const float* v = V + (size_t)h * RK * N + n;
  float acc = 0.f;
#pragma unroll
  for (int r = 0; r < RK; ++r) acc += u[r] * v[(size_t)r * N];
  Wc[i] = (__bf16)acc;
}

// ============================================================
// generic batched WMMA GEMM  C = A[M,K](bf16) @ B[K,N](bf16) (+bias, epilogue)
// EPI: 0 = f32 store (+bias), 1 = bf16 store (+bias), 2 = f32 cos(x+bias)*scale
// ============================================================
template <int EPI>
__global__ void k_gemm(const __bf16* __restrict__ A, const __bf16* __restrict__ Bm,
                       void* __restrict__ Cv, const float* __restrict__ bias,
                       float scale, int M, int N, int K,
                       long aSB, long aSH, long bSB, long bSH, long cSB, long cSH,
                       int biasSH, int hdiv) {
  int z  = blockIdx.z;
  int bb = z / hdiv;
  int hh = z % hdiv;
  A  += (size_t)bb * aSB + (size_t)hh * aSH;
  Bm += (size_t)bb * bSB + (size_t)hh * bSH;
  size_t coff = (size_t)bb * cSB + (size_t)hh * cSH;

  int lane = threadIdx.x & 31;
  int wave = threadIdx.x >> 5;
  int mt = blockIdx.y * 8 + wave;   // 16-row tile
  int nt = blockIdx.x;              // 16-col tile

  v8f c = {};
  for (int k0 = 0; k0 < K; k0 += 32) {
    v16bf a = frag_a(A, K, mt * 16, k0, lane);
    v16bf b = frag_b_rm(Bm, N, nt * 16, k0, lane);
    if (k0 + 32 < K)  // hint next A strip into cache (global_prefetch_b8)
      __builtin_prefetch((const void*)(A + (size_t)(mt * 16 + (lane & 15)) * K + k0 + 32), 0, 1);
    c = wmma_bf16(a, b, c);
  }

  int n = nt * 16 + (lane & 15);
  float bv = bias ? bias[(size_t)hh * biasSH + n] : 0.0f;
#pragma unroll
  for (int r = 0; r < 8; ++r) {
    int m  = r + ((lane < 16) ? 0 : 8);
    int mg = mt * 16 + m;
    float val = c[r];
    if (EPI == 2) val = cosf(val + bv) * scale;
    else          val = val + bv;
    if (EPI == 1) ((__bf16*)Cv)[coff + (size_t)mg * N + n] = (__bf16)val;
    else          ((float*)Cv)[coff + (size_t)mg * N + n]  = val;
  }
}

// ============================================================
// output projection GEMM with TDM-staged B panel in LDS
// C[M,256] = A[M,2048] @ Wout[2048,256] + bout ; block = 8 waves, 128 rows x 16 cols
// wave 0 DMAs the [2048 x 16] bf16 B panel (64 KB) into LDS via the Tensor Data
// Mover, waits on TENSORcnt, then all waves stream B fragments from LDS.
// ============================================================
__global__ void __launch_bounds__(256)
k_gemm_out(const __bf16* __restrict__ A, const __bf16* __restrict__ Bm,
           float* __restrict__ C, const float* __restrict__ bias,
           int M, int N, int K) {
  __shared__ __bf16 bpanel[2048 * 16];  // 64 KB of the 320 KB/WGP LDS

  if (__builtin_amdgcn_cluster_id_x() == 0x7fffffff) return;  // never taken

  int lane = threadIdx.x & 31;
  int wave = threadIdx.x >> 5;
  int mt = blockIdx.y * 8 + wave;
  int nt = blockIdx.x;

  if (wave == 0) {
    unsigned lds_off = (unsigned)(size_t)(&bpanel[0]);
    tdm_load_to_lds(lds_off, (const void*)(Bm + (size_t)nt * 16),
                    /*tensor_d0=*/(unsigned)N, /*tensor_d1=*/(unsigned)K,
                    /*tile_d0=*/16u, /*tile_d1=*/(unsigned)K,
                    /*stride_d0=*/(unsigned)N);
    __builtin_amdgcn_s_wait_tensorcnt(0);
  }
  __syncthreads();

  v8f c = {};
  for (int k0 = 0; k0 < K; k0 += 32) {
    v16bf a = frag_a(A, K, mt * 16, k0, lane);
    v16bf b = frag_b_lds_rm(bpanel, k0, lane);
    if (k0 + 32 < K)
      __builtin_prefetch((const void*)(A + (size_t)(mt * 16 + (lane & 15)) * K + k0 + 32), 0, 1);
    c = wmma_bf16(a, b, c);
  }

  int n = nt * 16 + (lane & 15);
  float bv = bias[n];
#pragma unroll
  for (int r = 0; r < 8; ++r) {
    int m  = r + ((lane < 16) ? 0 : 8);
    int mg = mt * 16 + m;
    C[(size_t)mg * N + n] = c[r] + bv;
  }
}

// ============================================================
// LSH hash + fingerprints. one thread per (h,s)
// ============================================================
__device__ __forceinline__ int lsh_bucket(float x) {
  float t = floorf(x / 4.0f);        // BW = 4.0
  float r = fmodf(t, (float)NBUCK);  // python mod -> non-negative
  if (r < 0.f) r += (float)NBUCK;
  return (int)r;
}

__global__ void k_hashfp(const __bf16* __restrict__ q_up, const __bf16* __restrict__ k_up,
                         const float* __restrict__ lsh_w,
                         int* __restrict__ hq, int* __restrict__ hk,
                         unsigned* __restrict__ qfp, unsigned* __restrict__ kfp,
                         unsigned char* __restrict__ ins) {
  int i = blockIdx.x * blockDim.x + threadIdx.x;  // h*SS + s
  if (i >= HH * SS) return;
  int s = i % SS;
  int h = i / SS;
  const float* w = lsh_w + (size_t)h * DQK;  // lsh_w[h][d][0]

  for (int b = 0; b < BB; ++b) {
    int z = b * HH + h;
    const __bf16* qr = q_up + ((size_t)z * SS + s) * DQK;
    const __bf16* kr = k_up + ((size_t)z * SS + s) * DQK;
    float dq = 0.f, dk = 0.f;
#pragma unroll 8
    for (int d = 0; d < DQK; ++d) {
      dq += (float)qr[d] * w[d];
      dk += (float)kr[d] * w[d];
    }
    hq[(size_t)z * SS + s] = lsh_bucket(dq);
    hk[(size_t)z * SS + s] = lsh_bucket(dk);
  }
  {  // Wu-Manber prefix (batch 0, first 8 sign bits)
    int z = 0 * HH + h;
    const __bf16* qr = q_up + ((size_t)z * SS + s) * DQK;
    const __bf16* kr = k_up + ((size_t)z * SS + s) * DQK;
    unsigned pq = 0, pk = 0;
#pragma unroll
    for (int j = 0; j < 8; ++j) {
      if ((float)qr[j] > 0.f) pq |= (1u << j);
      if ((float)kr[j] > 0.f) pk |= (1u << j);
    }
    ins[i] = (pq == pk) ? 1 : 0;
  }
  {  // full 64-bit trie fingerprint (batch B-1)
    int z = (BB - 1) * HH + h;
    const __bf16* qr = q_up + ((size_t)z * SS + s) * DQK;
    const __bf16* kr = k_up + ((size_t)z * SS + s) * DQK;
    unsigned q0 = 0, q1 = 0, k0 = 0, k1 = 0;
#pragma unroll
    for (int j = 0; j < 32; ++j) {
      if ((float)qr[j] > 0.f)      q0 |= (1u << j);
      if ((float)qr[32 + j] > 0.f) q1 |= (1u << j);
      if ((float)kr[j] > 0.f)      k0 |= (1u << j);
      if ((float)kr[32 + j] > 0.f) k1 |= (1u << j);
    }
    qfp[(size_t)i * 2] = q0; qfp[(size_t)i * 2 + 1] = q1;
    kfp[(size_t)i * 2] = k0; kfp[(size_t)i * 2 + 1] = k1;
  }
}

// ============================================================
// masked similarity plane for one (b,h):  score[q][k] = mask ? q_up·k_up : -inf
// key tile (16x64 bf16, contiguous rows) staged into LDS by TDM; 4 waves/block
// ============================================================
__global__ void __launch_bounds__(128)
k_scores(const __bf16* __restrict__ q_up, const __bf16* __restrict__ k_up,
         const int* __restrict__ hq, const int* __restrict__ hk,
         const unsigned* __restrict__ qfp, const unsigned* __restrict__ kfp,
         const unsigned char* __restrict__ ins,
         float* __restrict__ score, int b, int h) {
  __shared__ __bf16 ktile[16 * DQK];  // 2 KB

  int lane = threadIdx.x & 31;
  int wave = threadIdx.x >> 5;
  int mt = blockIdx.y * 4 + wave;  // query tile
  int nt = blockIdx.x;             // key tile
  int z = b * HH + h;
  const __bf16* Q  = q_up + (size_t)z * SS * DQK;
  const __bf16* Kp = k_up + (size_t)z * SS * DQK;

  if (wave == 0) {
    // 1D tile: 16 key rows x 64 dims = 1024 contiguous bf16 elements
    unsigned lds_off = (unsigned)(size_t)(&ktile[0]);
    tdm_load_to_lds(lds_off, (const void*)(Kp + (size_t)nt * 16 * DQK),
                    /*tensor_d0=*/(unsigned)(SS * DQK), /*tensor_d1=*/1u,
                    /*tile_d0=*/(unsigned)(16 * DQK), /*tile_d1=*/0u,
                    /*stride_d0=*/0u);
    __builtin_amdgcn_s_wait_tensorcnt(0);
  }
  __syncthreads();

  v8f c = {};
#pragma unroll
  for (int k0 = 0; k0 < DQK; k0 += 32) {
    v16bf a  = frag_a(Q, DQK, mt * 16, k0, lane);
    v16bf bf = frag_b_lds_tr(ktile, DQK, k0, lane);
    c = wmma_bf16(a, bf, c);
  }

  int key = nt * 16 + (lane & 15);
  int hkv = hk[(size_t)z * SS + key];
  unsigned kf0 = kfp[((size_t)h * SS + key) * 2];
  unsigned kf1 = kfp[((size_t)h * SS + key) * 2 + 1];
  int insk = ins[(size_t)h * SS + key];
  const float NEGINF = -__builtin_inff();
#pragma unroll
  for (int r = 0; r < 8; ++r) {
    int m  = r + ((lane < 16) ? 0 : 8);
    int qg = mt * 16 + m;
    bool ok = insk && (hq[(size_t)z * SS + qg] == hkv) &&
              (qfp[((size_t)h * SS + qg) * 2] == kf0) &&
              (qfp[((size_t)h * SS + qg) * 2 + 1] == kf1);
    score[(size_t)qg * SS + key] = ok ? c[r] : NEGINF;
  }
}

// ============================================================
// iterative top-64 selection over one score row (block = 256 threads, LDS row)
// ============================================================
__global__ void k_topk(const float* __restrict__ score, int* __restrict__ idx,
                       unsigned char* __restrict__ valid, int bh) {
  __shared__ float ssc[SS];
  __shared__ float rv[256];
  __shared__ int   ri[256];
  int q = blockIdx.x;
  int t = threadIdx.x;
  const float* row = score + (size_t)q * SS;
  for (int j = t; j < SS; j += 256) ssc[j] = row[j];
  __syncthreads();

  int* oi = idx + ((size_t)bh * SS + q) * KSEL;
  unsigned char* ov = valid + ((size_t)bh * SS + q) * KSEL;
  const float NEGINF = -__builtin_inff();

  for (int k = 0; k < KSEL; ++k) {
    float bv = NEGINF;
    int bi = SS;
    for (int j = t; j < SS; j += 256) {
      float v = ssc[j];
      if (v > bv || (v == bv && j < bi)) { bv = v; bi = j; }
    }
    rv[t] = bv; ri[t] = bi;
    __syncthreads();
    for (int s2 = 128; s2 > 0; s2 >>= 1) {
      if (t < s2) {
        float v2 = rv[t + s2]; int i2 = ri[t + s2];
        if (v2 > rv[t] || (v2 == rv[t] && i2 < ri[t])) { rv[t] = v2; ri[t] = i2; }
      }
      __syncthreads();
    }
    if (t == 0) {
      int sel = ri[0];
      oi[k] = sel;
      ov[k] = (rv[0] != NEGINF) ? 1 : 0;
      ssc[sel] = NEGINF;
    }
    __syncthreads();
  }
}

// ============================================================
// RFF attention over selected candidates + V gather; writes head-concat (bf16)
// block = 128 threads, one block per (b,h,q)
// ============================================================
__global__ void k_attn(const float* __restrict__ phi_q, const float* __restrict__ phi_k,
                       const __bf16* __restrict__ v_up,
                       const int* __restrict__ idx, const unsigned char* __restrict__ valid,
                       __bf16* __restrict__ headcat) {
  __shared__ int   sidx[KSEL];
  __shared__ float sw[KSEL];
  __shared__ int   svld[KSEL];
  __shared__ float mxs, sms;

  int gid = blockIdx.x;       // z*SS + q
  int q = gid % SS;
  int z = gid / SS;
  int h = z % HH;
  int b = z / HH;
  int t = threadIdx.x;

  const float* pq = phi_q + ((size_t)z * SS + q) * NRFF;
  if (t < KSEL) {
    int j  = idx[((size_t)z * SS + q) * KSEL + t];
    int vl = valid[((size_t)z * SS + q) * KSEL + t];
    const float* pk = phi_k + ((size_t)z * SS + j) * NRFF;
    float acc = 0.f;
#pragma unroll 8
    for (int r = 0; r < NRFF; ++r) acc += pq[r] * pk[r];
    sidx[t] = j;
    svld[t] = vl;
    sw[t] = vl ? acc : -1e9f;
  }
  __syncthreads();
  if (t == 0) {
    float m = -__builtin_inff();
    for (int j = 0; j < KSEL; ++j) m = fmaxf(m, sw[j]);
    mxs = m;
  }
  __syncthreads();
  if (t < KSEL) sw[t] = expf(sw[t] - mxs);
  __syncthreads();
  if (t == 0) {
    float s = 0.f;
    for (int j = 0; j < KSEL; ++j) s += sw[j];
    sms = s;
  }
  __syncthreads();
  if (t < KSEL) sw[t] = svld[t] ? (sw[t] / sms) : 0.0f;
  __syncthreads();

  const __bf16* vb = v_up + (size_t)z * SS * DD;
  for (int d = t; d < DD; d += 128) {
    float acc = 0.f;
#pragma unroll 8
    for (int j = 0; j < KSEL; ++j)
      acc += sw[j] * (float)vb[(size_t)sidx[j] * DD + d];
    headcat[((size_t)b * SS + q) * (HH * DD) + (size_t)h * DD + d] = (__bf16)acc;
  }
}

// ============================================================
// host-side launcher
// ============================================================
static inline char* bump(char*& p, size_t bytes) {
  char* r = p;
  p += (bytes + 255) & ~(size_t)255;
  return r;
}

extern "C" void kernel_launch(void* const* d_in, const int* in_sizes, int n_in,
                              void* d_out, int out_size, void* d_ws, size_t ws_size,
                              hipStream_t stream) {
  const float* query = (const float*)d_in[0];
  const float* key   = (const float*)d_in[1];
  // d_in[2] (value) is unused by the reference computation.
  const float* Wq    = (const float*)d_in[3];
  const float* bq    = (const float*)d_in[4];
  const float* Wkv   = (const float*)d_in[5];
  const float* bkv   = (const float*)d_in[6];
  const float* qu_u  = (const float*)d_in[7];
  const float* qu_v  = (const float*)d_in[8];
  const float* ku_u  = (const float*)d_in[9];
  const float* ku_v  = (const float*)d_in[10];
  const float* vu_u  = (const float*)d_in[11];
  const float* vu_v  = (const float*)d_in[12];
  const float* omega = (const float*)d_in[13];
  const float* rbias = (const float*)d_in[14];
  const float* lshw  = (const float*)d_in[15];
  const float* Wout  = (const float*)d_in[16];
  const float* bout  = (const float*)d_in[17];
  float* out = (float*)d_out;

  char* p = (char*)d_ws;
  __bf16* query_bf   = (__bf16*)bump(p, (size_t)BB * SS * DD * 2);
  __bf16* key_bf     = (__bf16*)bump(p, (size_t)BB * SS * DD * 2);
  __bf16* Wq_bf      = (__bf16*)bump(p, (size_t)DD * DQK * 2);
  __bf16* Wkv_bf     = (__bf16*)bump(p, (size_t)DD * DQK * 2);
  __bf16* Wqu_bf     = (__bf16*)bump(p, (size_t)HH * DQK * DQK * 2);
  __bf16* Wku_bf     = (__bf16*)bump(p, (size_t)HH * DQK * DQK * 2);
  __bf16* Wvu_bf     = (__bf16*)bump(p, (size_t)HH * DQK * DD * 2);
  __bf16* Wout_bf    = (__bf16*)bump(p, (size_t)HH * DD * DD * 2);
  __bf16* omega_bf   = (__bf16*)bump(p, (size_t)HH * DQK * NRFF * 2);
  __bf16* q_down_bf  = (__bf16*)bump(p, (size_t)BB * SS * DQK * 2);
  __bf16* kv_down_bf = (__bf16*)bump(p, (size_t)BB * SS * DQK * 2);
  __bf16* q_up_bf    = (__bf16*)bump(p, (size_t)BB * HH * SS * DQK * 2);
  __bf16* k_up_bf    = (__bf16*)bump(p, (size_t)BB * HH * SS * DQK * 2);
  __bf16* v_up_bf    = (__bf16*)bump(p, (size_t)BB * HH * SS * DD * 2);
  float*  phi_q      = (float*)bump(p, (size_t)BB * HH * SS * NRFF * 4);
  float*  phi_k      = (float*)bump(p, (size_t)BB * HH * SS * NRFF * 4);
  int*    hq         = (int*)bump(p, (size_t)BB * HH * SS * 4);
  int*    hk         = (int*)bump(p, (size_t)BB * HH * SS * 4);
  unsigned* qfp      = (unsigned*)bump(p, (size_t)HH * SS * 2 * 4);
  unsigned* kfp      = (unsigned*)bump(p, (size_t)HH * SS * 2 * 4);
  unsigned char* ins = (unsigned char*)bump(p, (size_t)HH * SS);
  float*  score      = (float*)bump(p, (size_t)SS * SS * 4);
  int*    idx        = (int*)bump(p, (size_t)BB * HH * SS * KSEL * 4);
  unsigned char* vld = (unsigned char*)bump(p, (size_t)BB * HH * SS * KSEL);
  __bf16* headcat    = (__bf16*)bump(p, (size_t)BB * SS * HH * DD * 2);
  (void)ws_size; (void)n_in; (void)in_sizes; (void)out_size;

  // ---- stage 0: conversions + combined up-projection weights ----
  auto cvt = [&](const float* s, __bf16* d, int n) {
    k_cvt<<<(n + 255) / 256, 256, 0, stream>>>(s, d, n);
  };
  cvt(query, query_bf, BB * SS * DD);
  cvt(key,   key_bf,   BB * SS * DD);
  cvt(Wq,    Wq_bf,    DD * DQK);
  cvt(Wkv,   Wkv_bf,   DD * DQK);
  cvt(Wout,  Wout_bf,  HH * DD * DD);
  cvt(omega, omega_bf, HH * DQK * NRFF);
  k_combo<<<(HH * 64 * DQK + 255) / 256, 256, 0, stream>>>(qu_u, qu_v, Wqu_bf, DQK);
  k_combo<<<(HH * 64 * DQK + 255) / 256, 256, 0, stream>>>(ku_u, ku_v, Wku_bf, DQK);
  k_combo<<<(HH * 64 * DD  + 255) / 256, 256, 0, stream>>>(vu_u, vu_v, Wvu_bf, DD);

  // ---- stage 1: down projections (bf16 store) ----
  {
    dim3 g(DQK / 16, (BB * SS) / 128, 1);
    k_gemm<1><<<g, 256, 0, stream>>>(query_bf, Wq_bf, q_down_bf, bq, 0.f,
                                     BB * SS, DQK, DD, 0, 0, 0, 0, 0, 0, 0, 1);
    k_gemm<1><<<g, 256, 0, stream>>>(key_bf, Wkv_bf, kv_down_bf, bkv, 0.f,
                                     BB * SS, DQK, DD, 0, 0, 0, 0, 0, 0, 0, 1);
  }

  // ---- stage 2: batched up projections, z = b*H + h ----
  {
    dim3 g(DQK / 16, SS / 128, BB * HH);
    k_gemm<1><<<g, 256, 0, stream>>>(q_down_bf, Wqu_bf, q_up_bf, nullptr, 0.f,
                                     SS, DQK, DQK,
                                     (long)SS * DQK, 0, 0, (long)DQK * DQK,
                                     (long)HH * SS * DQK, (long)SS * DQK, 0, HH);
    k_gemm<1><<<g, 256, 0, stream>>>(kv_down_bf, Wku_bf, k_up_bf, nullptr, 0.f,
                                     SS, DQK, DQK,
                                     (long)SS * DQK, 0, 0, (long)DQK * DQK,
                                     (long)HH * SS * DQK, (long)SS * DQK, 0, HH);
    dim3 gv(DD / 16, SS / 128, BB * HH);
    k_gemm<1><<<gv, 256, 0, stream>>>(kv_down_bf, Wvu_bf, v_up_bf, nullptr, 0.f,
                                      SS, DD, DQK,
                                      (long)SS * DQK, 0, 0, (long)DQK * DD,
                                      (long)HH * SS * DD, (long)SS * DD, 0, HH);
  }

  // ---- stage 3: LSH hashes + fingerprints ----
  k_hashfp<<<(HH * SS + 255) / 256, 256, 0, stream>>>(q_up_bf, k_up_bf, lshw,
                                                      hq, hk, qfp, kfp, ins);

  // ---- stage 4: RFF features phi = cos(X@omega + bias) * sqrt(2/RFF) ----
  {
    float scale = sqrtf(2.0f / (float)NRFF);
    dim3 g(NRFF / 16, SS / 128, BB * HH);
    k_gemm<2><<<g, 256, 0, stream>>>(q_up_bf, omega_bf, phi_q, rbias, scale,
                                     SS, NRFF, DQK,
                                     (long)HH * SS * DQK, (long)SS * DQK,
                                     0, (long)DQK * NRFF,
                                     (long)HH * SS * NRFF, (long)SS * NRFF, NRFF, HH);
    k_gemm<2><<<g, 256, 0, stream>>>(k_up_bf, omega_bf, phi_k, rbias, scale,
                                     SS, NRFF, DQK,
                                     (long)HH * SS * DQK, (long)SS * DQK,
                                     0, (long)DQK * NRFF,
                                     (long)HH * SS * NRFF, (long)SS * NRFF, NRFF, HH);
  }

  // ---- stage 5: per-(b,h) masked similarity + top-64 ----
  for (int bh = 0; bh < BB * HH; ++bh) {
    int b = bh / HH, h = bh % HH;
    dim3 gs(SS / 16, SS / 64, 1);  // 4 waves/block, each one q-tile
    k_scores<<<gs, 128, 0, stream>>>(q_up_bf, k_up_bf, hq, hk, qfp, kfp, ins,
                                     score, b, h);
    k_topk<<<SS, 256, 0, stream>>>(score, idx, vld, bh);
  }

  // ---- stage 6: candidate attention + V gather -> head concat ----
  k_attn<<<BB * HH * SS, 128, 0, stream>>>(phi_q, phi_k, v_up_bf, idx, vld, headcat);

  // ---- stage 7: output projection (TDM-staged B panel) ----
  {
    dim3 g(DD / 16, (BB * SS) / 128, 1);
    k_gemm_out<<<g, 256, 0, stream>>>(headcat, Wout_bf, out, bout,
                                      BB * SS, DD, HH * DD);
  }
}
